// decoder0_safenet_56607668961953
// MI455X (gfx1250) — matrix operand order repaired and checked by reading, not compiled, not run on hardware
//
#include <hip/hip_runtime.h>
#include <hip/hip_bf16.h>
#include <stdint.h>

// ---------------------------------------------------------------------------
// CDNA5 (gfx1250) WMMA helpers
// ---------------------------------------------------------------------------
typedef __attribute__((ext_vector_type(16))) _Float16 v16h;
typedef __attribute__((ext_vector_type(8)))  _Float16 v8h;
typedef __attribute__((ext_vector_type(8)))  float    v8f;
typedef __attribute__((ext_vector_type(4)))  float    v4f;

static __device__ __forceinline__ v8f v8zero() {
  v8f z = {0.f, 0.f, 0.f, 0.f, 0.f, 0.f, 0.f, 0.f};
  return z;
}

static __device__ __forceinline__ v8f wmma16(v16h a, v16h b, v8f c) {
  // D = A(16x32 f16) * B(32x16 f16) + C(16x16 f32)
  return __builtin_amdgcn_wmma_f32_16x16x32_f16(
      /*neg_a=*/false, a, /*neg_b=*/false, b,
      /*c_mod=*/(short)0, c, /*reuse_a=*/false, /*reuse_b=*/false);
}

// A fragment (16x32, f16). `row` points at the 32 contiguous halves of row
// M = (lane&15). Lanes 0-15 hold K=0..15; lanes 16-31 hold K=16..31.
static __device__ __forceinline__ v16h frag_a(const _Float16* row, int half) {
  return *(const v16h*)(row + half * 16);
}

// B fragment (32x16, f16) from K-contiguous column storage. `col` points at the
// 32 contiguous halves (K=0..31) of column N = (lane&15).
// Lanes 0-15 hold K=0..15, lanes 16-31 hold K=16..31.
static __device__ __forceinline__ v16h frag_b(const _Float16* col, int half) {
  return *(const v16h*)(col + half * 16);
}

// ---------------------------------------------------------------------------
// Implicit-GEMM convolution (stride 1).  M=Cout, N=H*W (per image), K=Cin*KH*KW.
// Block: 256 threads (8 waves), tile 64(M) x 128(N), BK=32.
// Wave (2x4 grid) owns a 32x32 sub-tile = 2x2 WMMA accumulators.
// KH/KW/pad are compile-time so the im2col K-decompose is division-free and the
// gather loop fully unrolls (batched loads, no per-element loadcnt stall).
// ---------------------------------------------------------------------------
template <int KH, int KW, int PAD>
__global__ __launch_bounds__(256)
void k_conv_gemm(const float* __restrict__ in, const float* __restrict__ w,
                 const float* __restrict__ bias, float* __restrict__ out,
                 int Cin, int Cout, int H, int W) {
  constexpr int KHW = KH * KW;
  __shared__ _Float16 As[64 * 32] __attribute__((aligned(64)));
  __shared__ _Float16 Bs[128 * 32] __attribute__((aligned(64)));

  const int HW   = H * W;
  const int Ktot = Cin * KHW;
  const int n0   = blockIdx.x * 128;
  const int m0   = blockIdx.y * 64;
  const int nimg = blockIdx.z;

  const int tid  = threadIdx.x;
  const int wave = tid >> 5;
  const int lane = tid & 31;
  const int half = lane >> 4;
  const int l15  = lane & 15;
  const int wm   = (wave >> 2) * 32;  // wave row offset within 64
  const int wn   = (wave & 3) * 32;   // wave col offset within 128

  const float* inN = in + (size_t)nimg * Cin * HW;

  // A staging: thread owns 8 contiguous K of one row (row = tid>>2).
  const int arow = tid >> 2;
  const int akc  = (tid & 3) * 8;
  const int acout = m0 + arow;
  const bool aok = acout < Cout;
  const float* awp = w + (size_t)(aok ? acout : 0) * Ktot + akc;

  // B staging: thread owns 16 contiguous K of one pixel (n = tid>>1).
  const int bn = tid >> 1;
  const int bkc = (tid & 1) * 16;
  const int bp = n0 + bn;
  const bool bpok = bp < HW;
  const int bpp = bpok ? bp : 0;
  const int bpy = bpp / W;
  const int bpx = bpp - bpy * W;

  v8f acc[2][2];
  for (int i = 0; i < 2; ++i)
    for (int j = 0; j < 2; ++j) acc[i][j] = v8zero();

  for (int k0 = 0; k0 < Ktot; k0 += 32) {
    // ---- stage A (weights): contiguous in K for OIHW -> 2x b128 + 1x ds_b128
    {
      v4f w0 = *(const v4f*)(awp + k0);
      v4f w1 = *(const v4f*)(awp + k0 + 4);
      v8h h;
      #pragma unroll
      for (int t = 0; t < 4; ++t) {
        h[t]     = (_Float16)(aok ? w0[t] : 0.f);
        h[t + 4] = (_Float16)(aok ? w1[t] : 0.f);
      }
      *(v8h*)(As + arow * 32 + akc) = h;
    }
    // ---- stage B (im2col gather): branch-free, batched loads ----
    {
      float bv[16];
      bool ok[16];
      #pragma unroll
      for (int t = 0; t < 16; ++t) {
        int kk  = k0 + bkc + t;
        int cin = kk / KHW;          // compile-time divisor
        int rs  = kk - cin * KHW;
        int r   = rs / KW;
        int s   = rs - r * KW;
        int y   = bpy + r - PAD;
        int x   = bpx + s - PAD;
        bool in_range = bpok & ((unsigned)y < (unsigned)H) &
                        ((unsigned)x < (unsigned)W);
        int yc = min(max(y, 0), H - 1);
        int xc = min(max(x, 0), W - 1);
        bv[t] = inN[(size_t)cin * HW + (size_t)yc * W + xc];
        ok[t] = in_range;
      }
      v16h hb;
      #pragma unroll
      for (int t = 0; t < 16; ++t) hb[t] = (_Float16)(ok[t] ? bv[t] : 0.f);
      *(v16h*)(Bs + bn * 32 + bkc) = hb;
    }
    __syncthreads();

    v16h a0 = frag_a(As + (wm + l15) * 32, half);
    v16h a1 = frag_a(As + (wm + 16 + l15) * 32, half);
    v16h b0 = frag_b(Bs + (wn + l15) * 32, half);
    v16h b1 = frag_b(Bs + (wn + 16 + l15) * 32, half);

    acc[0][0] = wmma16(a0, b0, acc[0][0]);
    acc[0][1] = wmma16(a0, b1, acc[0][1]);
    acc[1][0] = wmma16(a1, b0, acc[1][0]);
    acc[1][1] = wmma16(a1, b1, acc[1][1]);
    __syncthreads();
  }

  float* outN = out + (size_t)nimg * Cout * HW;
  for (int mi = 0; mi < 2; ++mi)
    for (int ni = 0; ni < 2; ++ni)
      for (int j = 0; j < 8; ++j) {
        int row = m0 + wm + mi * 16 + j + 8 * half;  // cout
        int col = n0 + wn + ni * 16 + l15;           // spatial
        if (row < Cout && col < HW)
          outN[(size_t)row * HW + col] = acc[mi][ni][j] + bias[row];
      }
}

// ---------------------------------------------------------------------------
// GroupNorm + ReLU, writing into a concat destination at channel offset dstC0.
// One block per (n, group).
// ---------------------------------------------------------------------------
__global__ __launch_bounds__(256)
void k_groupnorm_relu(const float* __restrict__ x, const float* __restrict__ gamma,
                      const float* __restrict__ beta, float* __restrict__ dst,
                      int C, int HW, int groups, int dstC0, int dstC) {
  __shared__ float rs[256], rq[256];
  const int n = blockIdx.x / groups;
  const int g = blockIdx.x - n * groups;
  const int Cg = C / groups;
  const int cnt = Cg * HW;
  const float* base = x + ((size_t)n * C + (size_t)g * Cg) * HW;
  const int tid = threadIdx.x;

  float s = 0.f, q = 0.f;
  for (int i = tid; i < cnt; i += 256) {
    float v = base[i];
    s += v;
    q += v * v;
  }
  rs[tid] = s; rq[tid] = q;
  __syncthreads();
  for (int o = 128; o > 0; o >>= 1) {
    if (tid < o) { rs[tid] += rs[tid + o]; rq[tid] += rq[tid + o]; }
    __syncthreads();
  }
  float mean = rs[0] / (float)cnt;
  float var  = rq[0] / (float)cnt - mean * mean;
  float inv  = rsqrtf(var + 1e-5f);

  for (int i = tid; i < cnt; i += 256) {
    int cl = i / HW;
    int p  = i - cl * HW;
    int c  = g * Cg + cl;
    float v = (base[i] - mean) * inv * gamma[c] + beta[c];
    v = fmaxf(v, 0.f);
    dst[((size_t)n * dstC + dstC0 + c) * HW + p] = v;
  }
}

// ---------------------------------------------------------------------------
// Channel-offset copy (for skip concats).
// ---------------------------------------------------------------------------
__global__ void k_copy_concat(const float* __restrict__ src, float* __restrict__ dst,
                              int C, int HW, int dstC0, int dstC, int total) {
  int i = blockIdx.x * blockDim.x + threadIdx.x;
  if (i >= total) return;
  int p = i % HW;
  int c = (i / HW) % C;
  int n = i / (C * HW);
  dst[((size_t)n * dstC + dstC0 + c) * HW + p] = src[i];
}

// ---------------------------------------------------------------------------
// Bilinear resize (half-pixel centers) with channel-offset destination.
// Covers both up2 and to_hw.
// ---------------------------------------------------------------------------
__global__ void k_resize_bilinear(const float* __restrict__ src, float* __restrict__ dst,
                                  int C, int Hi, int Wi, int Ho, int Wo,
                                  int dstC0, int dstC, int total) {
  int i = blockIdx.x * blockDim.x + threadIdx.x;
  if (i >= total) return;
  int HWo = Ho * Wo;
  int p = i % HWo;
  int c = (i / HWo) % C;
  int n = i / (C * HWo);
  int oy = p / Wo, ox = p - oy * Wo;

  float fy = ((float)oy + 0.5f) * (float)Hi / (float)Ho - 0.5f;
  float fx = ((float)ox + 0.5f) * (float)Wi / (float)Wo - 0.5f;
  fy = fminf(fmaxf(fy, 0.f), (float)(Hi - 1));
  fx = fminf(fmaxf(fx, 0.f), (float)(Wi - 1));
  int y0 = (int)fy, x0 = (int)fx;
  int y1 = min(y0 + 1, Hi - 1), x1 = min(x0 + 1, Wi - 1);
  float wy = fy - (float)y0, wx = fx - (float)x0;

  const float* b = src + ((size_t)n * C + c) * Hi * Wi;
  float v00 = b[(size_t)y0 * Wi + x0], v01 = b[(size_t)y0 * Wi + x1];
  float v10 = b[(size_t)y1 * Wi + x0], v11 = b[(size_t)y1 * Wi + x1];
  float v = (1.f - wy) * ((1.f - wx) * v00 + wx * v01) +
            wy * ((1.f - wx) * v10 + wx * v11);
  dst[((size_t)n * dstC + dstC0 + c) * HWo + p] = v;
}

// ---------------------------------------------------------------------------
// Build normalized embedding e16 [N][4800][32] (f16, 24 real dims, K-padded).
// emb: [N,24,240,320] bilinear->60x80, mask: [N,1,240,320] nearest->60x80,
// emb *= (2*mask-1), L2-normalize over channels.
// ---------------------------------------------------------------------------
__global__ void k_make_e16(const float* __restrict__ emb, const float* __restrict__ mask,
                           _Float16* __restrict__ e16, int total) {
  int i = blockIdx.x * blockDim.x + threadIdx.x;
  if (i >= total) return;
  int n = i / 4800;
  int p = i - n * 4800;
  int oy = p / 80, ox = p - oy * 80;

  float fy = ((float)oy + 0.5f) * 4.f - 0.5f;
  float fx = ((float)ox + 0.5f) * 4.f - 0.5f;
  fy = fminf(fmaxf(fy, 0.f), 239.f);
  fx = fminf(fmaxf(fx, 0.f), 319.f);
  int y0 = (int)fy, x0 = (int)fx;
  int y1 = min(y0 + 1, 239), x1 = min(x0 + 1, 319);
  float wy = fy - (float)y0, wx = fx - (float)x0;

  int my = min(239, (int)(((float)oy + 0.5f) * 4.f));
  int mx = min(319, (int)(((float)ox + 0.5f) * 4.f));
  float f = 2.f * mask[((size_t)n * 240 + my) * 320 + mx] - 1.f;

  float v[24];
  float s = 0.f;
  #pragma unroll 4
  for (int c = 0; c < 24; ++c) {
    const float* b = emb + ((size_t)n * 24 + c) * 240 * 320;
    float v00 = b[(size_t)y0 * 320 + x0], v01 = b[(size_t)y0 * 320 + x1];
    float v10 = b[(size_t)y1 * 320 + x0], v11 = b[(size_t)y1 * 320 + x1];
    float vv = (1.f - wy) * ((1.f - wx) * v00 + wx * v01) +
               wy * ((1.f - wx) * v10 + wx * v11);
    vv *= f;
    v[c] = vv;
    s += vv * vv;
  }
  float inv = 1.f / fmaxf(sqrtf(s), 1e-12f);
  _Float16* o = e16 + (size_t)i * 32;
  #pragma unroll
  for (int c = 0; c < 24; ++c) o[c] = (_Float16)(v[c] * inv);
  #pragma unroll
  for (int c = 24; c < 32; ++c) o[c] = (_Float16)0.f;
}

__global__ void k_f32_to_f16(const float* __restrict__ src, _Float16* __restrict__ dst,
                             int total) {
  int i = blockIdx.x * blockDim.x + threadIdx.x;
  if (i < total) dst[i] = (_Float16)src[i];
}

// ---------------------------------------------------------------------------
// Flash-attention affinity propagation:
//   A = softmax(e e^T), y = A q.    e16:[N][4800][32]  q16:[N][256][4800]
// Block: 8 waves; wave owns 16 queries x 128 channels; streams 32 keys/iter.
//
// Key simplification: e rows are L2-normalized, so every logit is in [-1,1].
// Softmax is shift-invariant, so no running max is needed: P = exp(S) directly
// (range [0.37, 2.72]; row sums over 4800 keys stay < 2^14, f32-safe).
// The row sum is computed BY THE MATRIX UNIT: lacc = P @ ones(32x16), leaving
// the row sum replicated across all 16 columns — zero cross-lane shuffles.
// Per key-block: 2 score WMMAs + 1 ones-WMMA + 8 PV WMMAs.
// ---------------------------------------------------------------------------
__global__ __launch_bounds__(256)
void k_attention(const _Float16* __restrict__ e16, const _Float16* __restrict__ q16,
                 float* __restrict__ y, int Npix, int C) {
  __shared__ _Float16 P[8][16 * 32] __attribute__((aligned(64)));

  const int nimg = blockIdx.y;
  const int tid  = threadIdx.x;
  const int wave = tid >> 5;
  const int lane = tid & 31;
  const int half = lane >> 4;
  const int l15  = lane & 15;
  const int qt   = blockIdx.x * 4 + (wave >> 1);  // query tile (16 rows)
  const int ch0  = (wave & 1) * 128;              // channel half
  if (qt * 16 >= Npix) return;                    // wave-uniform

  const _Float16* eN = e16 + (size_t)nimg * Npix * 32;
  const _Float16* qN = q16 + (size_t)nimg * C * Npix;
  _Float16* Pw = P[wave];

  v16h aQ = frag_a(eN + ((size_t)qt * 16 + l15) * 32, half);

  v16h onesf;
  #pragma unroll
  for (int t = 0; t < 16; ++t) onesf[t] = (_Float16)1.f;

  v8f lacc = v8zero();   // row-sum accumulator (replicated across columns)
  v8f acc[8];
  #pragma unroll
  for (int t = 0; t < 8; ++t) acc[t] = v8zero();

  const int nkb = Npix / 32;  // 150
  for (int kb = 0; kb < nkb; ++kb) {
    const _Float16* kbase = eN + (size_t)kb * 32 * 32;
    if (kb + 1 < nkb) __builtin_prefetch(kbase + 32 * 32, 0, 0);

    // scores: S(16q x 32k) = e_q (16x32) @ e_k^T (32x16) twice
    v16h b0 = frag_b(kbase + (size_t)l15 * 32, half);
    v16h b1 = frag_b(kbase + (size_t)(16 + l15) * 32, half);
    v8f s0 = wmma16(aQ, b0, v8zero());
    v8f s1 = wmma16(aQ, b1, v8zero());

    // P = exp(S); C-layout row = j + 8*half, col = l15
    #pragma unroll
    for (int j = 0; j < 8; ++j) {
      int row = j + 8 * half;
      Pw[row * 32 + l15]      = (_Float16)__expf(s0[j]);
      Pw[row * 32 + 16 + l15] = (_Float16)__expf(s1[j]);
    }
    asm volatile("s_wait_dscnt 0" ::: "memory");  // P stores -> A-frag loads (same wave)

    v16h pf = frag_a(Pw + l15 * 32, half);
    lacc = wmma16(pf, onesf, lacc);               // row sums via matrix unit

    const _Float16* vbase = qN + (size_t)kb * 32;
    #pragma unroll
    for (int t2 = 0; t2 < 8; ++t2) {
      int ch = ch0 + t2 * 16 + l15;
      v16h vf = frag_b(vbase + (size_t)ch * Npix, half);
      acc[t2] = wmma16(pf, vf, acc[t2]);
    }
  }

  float* yN = y + (size_t)nimg * C * Npix;
  #pragma unroll
  for (int j = 0; j < 8; ++j) {
    float inv = 1.f / fmaxf(lacc[j], 1e-30f);     // every column holds the row sum
    int q = qt * 16 + j + 8 * half;
    #pragma unroll
    for (int t2 = 0; t2 < 8; ++t2) {
      int ch = ch0 + t2 * 16 + l15;
      yN[(size_t)ch * Npix + q] = acc[t2][j] * inv;
    }
  }
}

// ---------------------------------------------------------------------------
// Batch-norm (training stats over N,H,W) + residual add, in place into xin4.
// One block per channel.
// ---------------------------------------------------------------------------
__global__ __launch_bounds__(256)
void k_bn_residual(const float* __restrict__ y2, const float* __restrict__ g,
                   const float* __restrict__ b, float* __restrict__ xin,
                   int N, int C, int HW) {
  __shared__ float rs[256], rq[256];
  const int c = blockIdx.x;
  const int tid = threadIdx.x;
  const int cnt = N * HW;

  float s = 0.f, q = 0.f;
  for (int i = tid; i < cnt; i += 256) {
    int n = i / HW, p = i - n * HW;
    float v = y2[((size_t)n * C + c) * HW + p];
    s += v; q += v * v;
  }
  rs[tid] = s; rq[tid] = q;
  __syncthreads();
  for (int o = 128; o > 0; o >>= 1) {
    if (tid < o) { rs[tid] += rs[tid + o]; rq[tid] += rq[tid + o]; }
    __syncthreads();
  }
  float mean = rs[0] / (float)cnt;
  float var  = rq[0] / (float)cnt - mean * mean;
  float inv  = rsqrtf(var + 1e-5f);

  for (int i = tid; i < cnt; i += 256) {
    int n = i / HW, p = i - n * HW;
    size_t idx = ((size_t)n * C + c) * HW + p;
    xin[idx] += g[c] * (y2[idx] - mean) * inv + b[c];
  }
}

// ---------------------------------------------------------------------------
// Final 3x3 conv, Cout=3, replication padding (coordinate clamp).
// ---------------------------------------------------------------------------
__global__ void k_final_conv(const float* __restrict__ x, const float* __restrict__ w,
                             const float* __restrict__ bias, float* __restrict__ out,
                             int H, int W, int total) {
  int i = blockIdx.x * blockDim.x + threadIdx.x;
  if (i >= total) return;
  int HW = H * W;
  int p  = i % HW;
  int co = (i / HW) % 3;
  int n  = i / (3 * HW);
  int oy = p / W, ox = p - oy * W;

  float s = bias[co];
  for (int ci = 0; ci < 64; ++ci) {
    const float* xb = x + ((size_t)n * 64 + ci) * HW;
    const float* wb = w + ((size_t)co * 64 + ci) * 9;
    #pragma unroll
    for (int r = 0; r < 3; ++r) {
      int yy = min(max(oy + r - 1, 0), H - 1);
      #pragma unroll
      for (int c = 0; c < 3; ++c) {
        int xx = min(max(ox + c - 1, 0), W - 1);
        s += xb[(size_t)yy * W + xx] * wb[r * 3 + c];
      }
    }
  }
  out[i] = s;
}

// ---------------------------------------------------------------------------
// Host side
// ---------------------------------------------------------------------------
extern "C" void kernel_launch(void* const* d_in, const int* in_sizes, int n_in,
                              void* d_out, int out_size, void* d_ws, size_t ws_size,
                              hipStream_t stream) {
  (void)in_sizes; (void)n_in; (void)out_size; (void)ws_size;
  const int N = 4;

  // ---- inputs (setup_inputs dict order, tuples flattened) ----
  const float* x1   = (const float*)d_in[1];
  const float* x2   = (const float*)d_in[2];
  const float* x3   = (const float*)d_in[3];
  const float* x4   = (const float*)d_in[4];
  const float* x5   = (const float*)d_in[5];
  const float* x6   = (const float*)d_in[6];
  const float* emb  = (const float*)d_in[7];
  const float* mask = (const float*)d_in[8];
  const float* cw[6], *cb[6], *gg[6], *gb[6];
  for (int i = 0; i < 6; ++i) {
    cw[i] = (const float*)d_in[9 + i];
    cb[i] = (const float*)d_in[15 + i];
    gg[i] = (const float*)d_in[21 + i];
    gb[i] = (const float*)d_in[27 + i];
  }
  const float* Wa1 = (const float*)d_in[33];
  const float* ba1 = (const float*)d_in[34];
  const float* Wa2 = (const float*)d_in[35];
  const float* ba2 = (const float*)d_in[36];
  const float* bng = (const float*)d_in[37];
  const float* bnb = (const float*)d_in[38];
  const float* fw  = (const float*)d_in[39];
  const float* fb  = (const float*)d_in[40];
  float* out = (float*)d_out;

  // ---- workspace slots (disjoint lifetimes share regions) ----
  char* ws = (char*)d_ws;
  const size_t MBy = 1024 * 1024;
  float*     slotA = (float*)(ws + 0 * MBy);        // up buffers u1..u5 (<=158MB)
  float*     slotB = (float*)(ws + 160 * MBy);      // raw conv outputs (<=79MB)
  float*     slotC = (float*)(ws + 240 * MBy);      // xin1..xin5, dx6 (<=79MB)
  float*     slotD = (float*)(ws + 320 * MBy);      // dx2gn, q, yattn (<=20MB)
  float*     slotE = (float*)(ws + 340 * MBy);      // y2 (<=20MB)
  _Float16*  e16   = (_Float16*)(ws + 360 * MBy);   // 1.2MB
  _Float16*  q16   = (_Float16*)(ws + 362 * MBy);   // 9.8MB

  auto cdiv = [](int a, int b) { return (a + b - 1) / b; };

  auto conv3 = [&](const float* in, int Cin, int H, int W, const float* w,
                   const float* b, int Cout, float* o) {
    dim3 g(cdiv(H * W, 128), cdiv(Cout, 64), N);
    k_conv_gemm<3, 3, 1><<<g, 256, 0, stream>>>(in, w, b, o, Cin, Cout, H, W);
  };
  auto conv1 = [&](const float* in, int Cin, int H, int W, const float* w,
                   const float* b, int Cout, float* o) {
    dim3 g(cdiv(H * W, 128), cdiv(Cout, 64), N);
    k_conv_gemm<1, 1, 0><<<g, 256, 0, stream>>>(in, w, b, o, Cin, Cout, H, W);
  };
  auto gn = [&](const float* x, int C, int HW, int groups, const float* ga,
                const float* be, float* dst, int dstC0, int dstC) {
    k_groupnorm_relu<<<N * groups, 256, 0, stream>>>(x, ga, be, dst, C, HW, groups,
                                                     dstC0, dstC);
  };
  auto copyc = [&](const float* src, int C, int HW, float* dst, int dstC0, int dstC) {
    int total = N * C * HW;
    k_copy_concat<<<cdiv(total, 256), 256, 0, stream>>>(src, dst, C, HW, dstC0, dstC,
                                                        total);
  };
  auto resize = [&](const float* src, int C, int Hi, int Wi, int Ho, int Wo,
                    float* dst, int dstC0, int dstC) {
    int total = N * C * Ho * Wo;
    k_resize_bilinear<<<cdiv(total, 256), 256, 0, stream>>>(src, dst, C, Hi, Wi, Ho,
                                                            Wo, dstC0, dstC, total);
  };

  // ---- stage 1: x6 -> dx1 -> xin1 (1024ch, 8x10) ----
  conv3(x6, 1024, 8, 10, cw[0], cb[0], 512, slotB);
  gn(slotB, 512, 80, 32, gg[0], gb[0], slotC, 0, 1024);
  copyc(x5, 512, 80, slotC, 512, 1024);

  // ---- stage 2: up2 -> conv2 -> GN -> resize(16->15) -> xin2 (512ch,15x20) ----
  resize(slotC, 1024, 8, 10, 16, 20, slotA, 0, 1024);            // u1
  conv3(slotA, 1024, 16, 20, cw[1], cb[1], 256, slotB);          // dx2 raw
  gn(slotB, 256, 320, 16, gg[1], gb[1], slotD, 0, 256);          // dx2gn (16x20)
  resize(slotD, 256, 16, 20, 15, 20, slotC, 0, 512);             // -> xin2[0:256]
  copyc(x4, 256, 300, slotC, 256, 512);

  // ---- stage 3: up2 -> conv3 -> GN -> xin3 (512ch, 30x40) ----
  resize(slotC, 512, 15, 20, 30, 40, slotA, 0, 512);             // u2
  conv3(slotA, 512, 30, 40, cw[2], cb[2], 256, slotB);
  gn(slotB, 256, 1200, 16, gg[2], gb[2], slotC, 0, 512);
  copyc(x3, 256, 1200, slotC, 256, 512);

  // ---- stage 4: up2 -> conv4 -> GN -> xin4 (256ch, 60x80) ----
  resize(slotC, 512, 30, 40, 60, 80, slotA, 0, 512);             // u3
  conv3(slotA, 512, 60, 80, cw[3], cb[3], 128, slotB);
  gn(slotB, 128, 4800, 8, gg[3], gb[3], slotC, 0, 256);
  copyc(x2, 128, 4800, slotC, 128, 256);

  // ---- affinity propagation at xin4 ----
  conv1(slotC, 256, 60, 80, Wa1, ba1, 256, slotD);               // q (f32)
  k_make_e16<<<cdiv(N * 4800, 256), 256, 0, stream>>>(emb, mask, e16, N * 4800);
  {
    int total = N * 256 * 4800;
    k_f32_to_f16<<<cdiv(total, 256), 256, 0, stream>>>(slotD, q16, total);
  }
  k_attention<<<dim3(4800 / 16 / 4, N), 256, 0, stream>>>(e16, q16, slotD, 4800, 256);
  conv1(slotD, 256, 60, 80, Wa2, ba2, 256, slotE);               // y2
  k_bn_residual<<<256, 256, 0, stream>>>(slotE, bng, bnb, slotC, N, 256, 4800);

  // ---- stage 5: up2 -> conv5 -> GN -> xin5 (128ch, 120x160) ----
  resize(slotC, 256, 60, 80, 120, 160, slotA, 0, 256);           // u4
  conv3(slotA, 256, 120, 160, cw[4], cb[4], 64, slotB);
  gn(slotB, 64, 19200, 4, gg[4], gb[4], slotC, 0, 128);
  copyc(x1, 64, 19200, slotC, 64, 128);

  // ---- stage 6: up2 -> conv6 -> GN -> dx6 (64ch, 240x320) ----
  resize(slotC, 128, 120, 160, 240, 320, slotA, 0, 128);         // u5
  conv3(slotA, 128, 240, 320, cw[5], cb[5], 64, slotB);
  gn(slotB, 64, 76800, 4, gg[5], gb[5], slotC, 0, 64);           // dx6

  // ---- final: replication-pad 3x3 conv -> out [4,3,240,320] ----
  {
    int total = N * 3 * 240 * 320;
    k_final_conv<<<cdiv(total, 256), 256, 0, stream>>>(slotC, fw, fb, out, 240, 320,
                                                       total);
  }
}
